// GATLayer_80161269613388
// MI455X (gfx1250) — matrix operand order repaired and checked by reading, not compiled, not run on hardware
//
#include <hip/hip_runtime.h>
#include <math.h>

typedef __attribute__((ext_vector_type(16))) __bf16 v16bf;
typedef __attribute__((ext_vector_type(8)))  float  v8f;

#define HD 256
#define NEG_SLOPE 0.2f

// ---------- helpers ----------
__device__ __forceinline__ unsigned short f2bf(float f) {
  unsigned int x = __float_as_uint(f);
  x += 0x7FFFu + ((x >> 16) & 1u);      // round-to-nearest-even
  return (unsigned short)(x >> 16);
}
// order-preserving float -> uint key (for atomicMax-based segment max)
__device__ __forceinline__ unsigned int fkey(float f) {
  unsigned int b = __float_as_uint(f);
  return (b & 0x80000000u) ? ~b : (b | 0x80000000u);
}
__device__ __forceinline__ float fdecode(unsigned int k) {
  unsigned int b = (k & 0x80000000u) ? (k & 0x7FFFFFFFu) : ~k;
  return __uint_as_float(b);
}

// ---------- prep: fp32 -> bf16 conversions ----------
__global__ void cvt_feats_kernel(const float* __restrict__ feats,
                                 unsigned short* __restrict__ fb, long long total) {
  long long t = (long long)blockIdx.x * blockDim.x + threadIdx.x;
  if (t < total) fb[t] = f2bf(feats[t]);
}

// combined weights: rows 0..255 = fc_w, rows 256..511 = res_w  ([512,128] bf16)
__global__ void cvt_w_kernel(const float* __restrict__ fc_w,
                             const float* __restrict__ res_w,
                             unsigned short* __restrict__ wb) {
  int t = blockIdx.x * blockDim.x + threadIdx.x;
  if (t >= 512 * 128) return;
  int row = t >> 7, k = t & 127;
  float v = (row < 256) ? fc_w[row * 128 + k] : res_w[(row - 256) * 128 + k];
  wb[t] = f2bf(v);
}

// ---------- re-init accumulators each call ----------
__global__ void init_kernel(unsigned int* __restrict__ mkey, float* __restrict__ ssum,
                            float* __restrict__ bnsum, float* __restrict__ bnsq, int NH) {
  int t = blockIdx.x * blockDim.x + threadIdx.x;
  if (t < NH) { mkey[t] = 0u; ssum[t] = 0.0f; }
  if (t < 64) { bnsum[t] = 0.0f; bnsq[t] = 0.0f; }
}

// ---------- fused projection GEMM via WMMA (bf16 -> f32) ----------
// D[N,512] = feats[N,128] x Wc^T ; cols 0..255 -> h, cols 256..511 -> agg(=res).
// One block per 16-row m-tile; wave w sweeps n-tiles 4w..4w+3 reusing its A
// registers (full K=128 row loaded once) -> 16 WMMAs per wave.
__global__ void __launch_bounds__(256)
gat_gemm_wmma(const unsigned short* __restrict__ fb,
              const unsigned short* __restrict__ wb,
              float* __restrict__ h, float* __restrict__ agg, int N) {
  const int lane = threadIdx.x & 31;
  const int wave = threadIdx.x >> 5;
  const int m0 = blockIdx.x * 16;
  const bool hi = lane >= 16;
  const int khalfA = hi ? 8 : 0;                 // A: K-halves split at lane 16
  const int kbB    = hi ? 8 : 0;                 // B: K-halves split at lane 16 (same as A)
  int m = m0 + (lane & 15);
  if (m >= N) m = N - 1;                         // clamp loads of ragged tail tile
  const unsigned short* arow = fb + (size_t)m * 128;

  union AB { v16bf v; unsigned int u[8]; };

  // Load the full A row (K=128) once: 4 k-steps x 8 packed bf16 pairs.
  AB Aks[4];
#pragma unroll
  for (int ks = 0; ks < 4; ++ks) {
    const int k0 = ks * 32;
#pragma unroll
    for (int r = 0; r < 8; ++r) {
      const int ka = k0 + ((r < 4) ? 0 : 16) + khalfA + 2 * (r & 3);
      Aks[ks].u[r] = *(const unsigned int*)(arow + ka);
    }
  }

  const bool fullTile = (m0 + 15) < N;
  const int rhalf = hi ? 8 : 0;                  // D rows i / i+8 split at lane 16

#pragma unroll
  for (int nt = 0; nt < 4; ++nt) {
    const int n0 = (wave * 4 + nt) * 16;
    const int ncol = n0 + (lane & 15);
    const unsigned short* brow = wb + (size_t)ncol * 128;

    v8f acc = {};
#pragma unroll
    for (int ks = 0; ks < 4; ++ks) {
      AB B;
      const int k0 = ks * 32 + kbB;
#pragma unroll
      for (int r = 0; r < 8; ++r)
        B.u[r] = *(const unsigned int*)(brow + k0 + ((r < 4) ? 0 : 16) + 2 * (r & 3));
      acc = __builtin_amdgcn_wmma_f32_16x16x32_bf16(false, Aks[ks].v, false, B.v,
                                                    (short)0, acc, false, false);
    }

    // single base pointer + constant offsets (i*HD floats = i*1024 bytes)
    float* base = (n0 < HD)
        ? (h   + (size_t)(m0 + rhalf) * HD + ncol)
        : (agg + (size_t)(m0 + rhalf) * HD + (ncol - HD));
    if (fullTile) {
#pragma unroll
      for (int i = 0; i < 8; ++i) base[i * HD] = acc[i];
    } else {
      const int rbase = m0 + rhalf;
#pragma unroll
      for (int i = 0; i < 8; ++i)
        if (rbase + i < N) base[i * HD] = acc[i];
    }
  }
}

// ---------- per-node attention logits el/er ----------
__global__ void logits_kernel(const float* __restrict__ h,
                              const float* __restrict__ attn_l,
                              const float* __restrict__ attn_r,
                              float* __restrict__ el, float* __restrict__ er, int N) {
  int t = blockIdx.x * blockDim.x + threadIdx.x;
  if (t >= N * 4) return;
  int n = t >> 2, head = t & 3;
  const float4* hp = (const float4*)(h + (size_t)n * HD + head * 64);
  const float4* lp = (const float4*)(attn_l + head * 64);
  const float4* rp = (const float4*)(attn_r + head * 64);
  float sl = 0.f, sr = 0.f;
#pragma unroll 4
  for (int i = 0; i < 16; ++i) {
    float4 hv = hp[i], lv = lp[i], rv = rp[i];
    sl += hv.x * lv.x + hv.y * lv.y + hv.z * lv.z + hv.w * lv.w;
    sr += hv.x * rv.x + hv.y * rv.y + hv.z * rv.z + hv.w * rv.w;
  }
  el[t] = sl; er[t] = sr;
}

// ---------- segment max over incoming edges (ordered-uint atomicMax) ----------
__global__ void edge_max_kernel(const long long* __restrict__ src,
                                const long long* __restrict__ dst,
                                const float* __restrict__ el, const float* __restrict__ er,
                                unsigned int* __restrict__ mkey, int E) {
  int t = blockIdx.x * blockDim.x + threadIdx.x;
  if (t >= E * 4) return;
  int e = t >> 2, head = t & 3;
  long long s = src[e], d = dst[e];
  float x = el[s * 4 + head] + er[d * 4 + head];
  x = (x > 0.f) ? x : NEG_SLOPE * x;
  atomicMax(&mkey[d * 4 + head], fkey(x));
}

// ---------- exp(e - max) and segment sum ----------
__global__ void edge_exp_kernel(const long long* __restrict__ src,
                                const long long* __restrict__ dst,
                                const float* __restrict__ el, const float* __restrict__ er,
                                const unsigned int* __restrict__ mkey,
                                float* __restrict__ exb, float* __restrict__ ssum, int E) {
  int t = blockIdx.x * blockDim.x + threadIdx.x;
  if (t >= E * 4) return;
  int e = t >> 2, head = t & 3;
  long long s = src[e], d = dst[e];
  float x = el[s * 4 + head] + er[d * 4 + head];
  x = (x > 0.f) ? x : NEG_SLOPE * x;
  float ev = __expf(x - fdecode(mkey[d * 4 + head]));
  exb[t] = ev;
  atomicAdd(&ssum[d * 4 + head], ev);
}

// ---------- weighted message scatter: agg[dst] += alpha * h[src] ----------
__global__ void edge_agg_kernel(const long long* __restrict__ src,
                                const long long* __restrict__ dst,
                                const float* __restrict__ exb, const float* __restrict__ ssum,
                                const float* __restrict__ h, float* __restrict__ agg,
                                long long E) {
  long long t = (long long)blockIdx.x * blockDim.x + threadIdx.x;
  if (t >= E * 64) return;                  // one float4 of the 256-wide row per thread
  long long e = t >> 6;
  int q = (int)(t & 63);
  int head = q >> 4;
  long long s = src[e], d = dst[e];
  float alpha = exb[e * 4 + head] / ssum[d * 4 + head];
  float4 hv = *(const float4*)(h + s * HD + q * 4);
  float* ap = agg + d * HD + q * 4;
  atomicAdd(ap + 0, alpha * hv.x);
  atomicAdd(ap + 1, alpha * hv.y);
  atomicAdd(ap + 2, alpha * hv.z);
  atomicAdd(ap + 3, alpha * hv.w);
}

// ---------- ELU + bias + head-mean + BN partial sums ----------
__global__ void finalize_kernel(const float* __restrict__ agg, const float* __restrict__ bias,
                                float* __restrict__ out,
                                float* __restrict__ bnsum, float* __restrict__ bnsq, int N) {
  __shared__ float s1[8][64];
  __shared__ float s2[8][64];
  const int d = threadIdx.x, ty = threadIdx.y;
  const float b0 = bias[d], b1 = bias[64 + d], b2 = bias[128 + d], b3 = bias[192 + d];
  float asum = 0.f, asq = 0.f;
  for (int n = blockIdx.x * 8 + ty; n < N; n += gridDim.x * 8) {
    const float* a = agg + (size_t)n * HD;
    float v = 0.f, x;
    x = a[d]       + b0; v += (x > 0.f) ? x : expm1f(x);
    x = a[64 + d]  + b1; v += (x > 0.f) ? x : expm1f(x);
    x = a[128 + d] + b2; v += (x > 0.f) ? x : expm1f(x);
    x = a[192 + d] + b3; v += (x > 0.f) ? x : expm1f(x);
    v *= 0.25f;
    out[(size_t)n * 64 + d] = v;
    asum += v; asq += v * v;
  }
  s1[ty][d] = asum; s2[ty][d] = asq;
  __syncthreads();
  if (ty == 0) {
    float t1 = 0.f, t2 = 0.f;
#pragma unroll
    for (int i = 0; i < 8; ++i) { t1 += s1[i][d]; t2 += s2[i][d]; }
    atomicAdd(&bnsum[d], t1);
    atomicAdd(&bnsq[d], t2);
  }
}

__global__ void bn_stats_kernel(const float* __restrict__ bnsum, const float* __restrict__ bnsq,
                                const float* __restrict__ gamma, const float* __restrict__ beta,
                                float* __restrict__ scale, float* __restrict__ shift, int N) {
  int d = threadIdx.x;
  float inv = 1.0f / (float)N;
  float mu = bnsum[d] * inv;
  float var = bnsq[d] * inv - mu * mu;
  float sc = gamma[d] * rsqrtf(var + 1e-5f);
  scale[d] = sc;
  shift[d] = beta[d] - mu * sc;
}

__global__ void bn_apply_kernel(float* __restrict__ out, const float* __restrict__ scale,
                                const float* __restrict__ shift, long long total) {
  long long t = (long long)blockIdx.x * blockDim.x + threadIdx.x;
  if (t >= total) return;
  int d = (int)(t & 63);
  out[t] = out[t] * scale[d] + shift[d];
}

// ---------- host launcher ----------
extern "C" void kernel_launch(void* const* d_in, const int* in_sizes, int n_in,
                              void* d_out, int out_size, void* d_ws, size_t ws_size,
                              hipStream_t stream) {
  const float*     feats  = (const float*)d_in[0];
  const long long* src    = (const long long*)d_in[1];   // int64 per reference
  const long long* dst    = (const long long*)d_in[2];
  const float*     fc_w   = (const float*)d_in[3];
  const float*     attn_l = (const float*)d_in[4];
  const float*     attn_r = (const float*)d_in[5];
  const float*     res_w  = (const float*)d_in[6];
  const float*     bias   = (const float*)d_in[7];
  const float*     gamma  = (const float*)d_in[8];
  const float*     beta   = (const float*)d_in[9];
  float* out = (float*)d_out;

  const int N = in_sizes[0] / 128;
  const int E = in_sizes[1];

  // workspace layout (256B aligned slots)
  char* ws = (char*)d_ws;
  size_t off = 0;
  auto take = [&](size_t bytes) -> char* {
    char* p = ws + off;
    off = (off + bytes + 255) & ~(size_t)255;
    return p;
  };
  float*          h_buf = (float*)take((size_t)N * HD * 4);       // projected feats
  float*          agg   = (float*)take((size_t)N * HD * 4);       // res -> res + sum(msg)
  unsigned short* fb    = (unsigned short*)take((size_t)N * 128 * 2);
  unsigned short* wb    = (unsigned short*)take((size_t)512 * 128 * 2);
  float*          el    = (float*)take((size_t)N * 4 * 4);
  float*          er    = (float*)take((size_t)N * 4 * 4);
  unsigned int*   mkey  = (unsigned int*)take((size_t)N * 4 * 4);
  float*          ssum  = (float*)take((size_t)N * 4 * 4);
  float*          exb   = (float*)take((size_t)E * 4 * 4);
  float*          bnsum = (float*)take(64 * 4);
  float*          bnsq  = (float*)take(64 * 4);
  float*          scale = (float*)take(64 * 4);
  float*          shift = (float*)take(64 * 4);
  (void)ws_size; (void)n_in; (void)out_size;

  const int B = 256;
  // 1. convert inputs to bf16
  {
    long long tot = (long long)N * 128;
    cvt_feats_kernel<<<(unsigned)((tot + B - 1) / B), B, 0, stream>>>(feats, fb, tot);
    cvt_w_kernel<<<(512 * 128 + B - 1) / B, B, 0, stream>>>(fc_w, res_w, wb);
  }
  // 2. zero/neg-inf accumulators (every call — replays don't re-poison)
  init_kernel<<<(N * 4 + B - 1) / B, B, 0, stream>>>(mkey, ssum, bnsum, bnsq, N * 4);
  // 3. fused projection GEMM on WMMA: h and agg(=res)
  gat_gemm_wmma<<<(N + 15) / 16, B, 0, stream>>>(fb, wb, h_buf, agg, N);
  // 4. per-node attention logits
  logits_kernel<<<(N * 4 + B - 1) / B, B, 0, stream>>>(h_buf, attn_l, attn_r, el, er, N);
  // 5-7. edge softmax + weighted scatter
  edge_max_kernel<<<(E * 4 + B - 1) / B, B, 0, stream>>>(src, dst, el, er, mkey, E);
  edge_exp_kernel<<<(E * 4 + B - 1) / B, B, 0, stream>>>(src, dst, el, er, mkey, exb, ssum, E);
  {
    long long tot = (long long)E * 64;
    edge_agg_kernel<<<(unsigned)((tot + B - 1) / B), B, 0, stream>>>(src, dst, exb, ssum,
                                                                     h_buf, agg, E);
  }
  // 8. ELU + head mean + BN partial sums
  finalize_kernel<<<512, dim3(64, 8), 0, stream>>>(agg, bias, out, bnsum, bnsq, N);
  // 9. BN stats + apply
  bn_stats_kernel<<<1, 64, 0, stream>>>(bnsum, bnsq, gamma, beta, scale, shift, N);
  {
    long long tot = (long long)N * 64;
    bn_apply_kernel<<<(unsigned)((tot + B - 1) / B), B, 0, stream>>>(out, scale, shift, tot);
  }
}
